// UserEncoder_rate_61203283968767
// MI455X (gfx1250) — compile-verified, matrix operand order checked by reading, not run on hardware
//
#include <hip/hip_runtime.h>

// ---------------------------------------------------------------------------
// MI455X (gfx1250) implementation.
//
// All GEMMs (~161 GFLOP) run on the matrix cores via V_WMMA_F32_16X16X32_BF16
// (A/B bf16, f32 accumulate). Weights are pre-packed into the exact WMMA
// B-fragment lane layout; per-workgroup LDS staging gives each 512KB weight a
// single L2 read per WG pass. B fragments are double-buffered in registers so
// ds_load latency overlaps the WMMA pipe instead of serializing on
// s_wait_dscnt 0.
//
// Input order assumption: harness flattens setup_inputs() in dict insertion
// order, nested dicts included:
//   0..5   item_attn  attn1.w/b attn2.w/b attn3.w/b
//   6..11  user_attn  attn1.w/b attn2.w/b attn3.w/b
//   12,13  rate_lin w/b   14,15 fuse_lin w/b  16,17 self_lin w/b
//   18,19  exp_lin w/b    20,21 unexp_lin w/b
//   22 user_embs  23 item_embs  24 rate_embs  25 user  26 user_hist  27 user_nbrs
// Output: [unexp_h (1024x256) | exp_h (1024x256)] f32.
// ---------------------------------------------------------------------------

typedef __bf16 bf16x16 __attribute__((ext_vector_type(16)));
typedef __bf16 bf16x8  __attribute__((ext_vector_type(8)));
typedef float  v8f     __attribute__((ext_vector_type(8)));

#define EDIM 256
#define BATCH 1024
#define HISTL 200
#define NNBR 64
#define NEGV -1000000000.0f

// ---- ws layout (bf16 elements / bytes) ----
#define PK_RATE   0
#define PK_A1I    131072
#define PK_A2I    262144
#define PK_A1U    327680
#define PK_A2U    458752
#define PK_FUSE   524288
#define PK_SELF   655360
#define PK_UNEXP  786432
#define PK_EXP    851968
#define WS_HITEM_BYTES  1835008           // = 917504 bf16 elements * 2
#define WS_HSOC_BYTES   (WS_HITEM_BYTES + 1048576)

__device__ __forceinline__ v8f wmma_bf16(bf16x16 a, bf16x16 b, v8f c) {
  return __builtin_amdgcn_wmma_f32_16x16x32_bf16(false, a, false, b,
                                                 (short)0, c, false, false);
}

__device__ __forceinline__ bf16x16 combine(bf16x8 lo, bf16x8 hi) {
  return __builtin_shufflevector(lo, hi, 0,1,2,3,4,5,6,7,8,9,10,11,12,13,14,15);
}

__device__ __forceinline__ void cvt8(bf16x16& a, int base, float4 f0, float4 f1) {
  a[base+0]=(__bf16)f0.x; a[base+1]=(__bf16)f0.y; a[base+2]=(__bf16)f0.z; a[base+3]=(__bf16)f0.w;
  a[base+4]=(__bf16)f1.x; a[base+5]=(__bf16)f1.y; a[base+6]=(__bf16)f1.z; a[base+7]=(__bf16)f1.w;
}

// 16 WMMAs over one staged weight slice, with B-fragment double buffering so
// the next ds_load pair is in flight while the current WMMA issues.
__device__ __forceinline__ void wmma_sweep16(const __bf16* Wbuf, int lane,
                                             bf16x16 a, v8f* c) {
  bf16x16 bb_cur = *(const bf16x16*)(Wbuf + lane*16);
#pragma unroll
  for (int nt = 0; nt < 16; ++nt) {
    bf16x16 bb_nxt = bb_cur;
    if (nt < 15) bb_nxt = *(const bf16x16*)(Wbuf + ((nt+1)*32 + lane)*16);
    c[nt] = wmma_bf16(a, bb_cur, c[nt]);
    bb_cur = bb_nxt;
  }
}

// cooperative copy of one 32-K-deep weight slice (all 16 N-tiles) into LDS.
// Packed layout chunk (nt,kb) = 32 lanes * 16 bf16 = 1KB = 64 uint4.
__device__ __forceinline__ void stage_w(__bf16* Wbuf, const __bf16* wpk,
                                        int kb, int KB, int tid) {
  uint4* dst = (uint4*)Wbuf;
  const uint4* src = (const uint4*)wpk;
#pragma unroll
  for (int ii = 0; ii < 4; ++ii) {
    int i = tid + ii*256;
    int c = i >> 6, j = i & 63;
    dst[i] = src[(size_t)(c*KB + kb)*64 + j];
  }
}

// ---------------------------------------------------------------------------
// Weight pack: f32 (K x N row-major) -> bf16 fragment-packed B layout.
// For V_WMMA_F32_16X16X32_BF16, lane L of a B fragment holds column n=L&15,
// K elements: lanes 0-15 -> k {0..7,16..23}, lanes 16-31 -> k {8..15,24..31}.
// ---------------------------------------------------------------------------
__global__ void pack_w_kernel(const float* __restrict__ src, __bf16* __restrict__ dst,
                              int K, int N) {
  int i = blockIdx.x*256 + threadIdx.x;
  if (i >= K*N) return;
  int k = i / N, n = i - k*N;
  int nt = n >> 4, kb = k >> 5, ko = k & 31;
  int hi = (ko >> 3) & 1;
  int e = (ko & 7) + ((ko >> 4) << 3);
  int lane = (n & 15) + (hi << 4);
  dst[((size_t)((nt*(K>>5) + kb)*32 + lane) << 4) + e] = (__bf16)src[i];
}

// ---------------------------------------------------------------------------
// Fused attention branch. 1 workgroup per batch row, 256 threads (8 waves).
// mode 0 (item): E = rate_lin([item_emb|rate_emb]); mode 1 (social): E = gather.
// Then H1 = leaky([E|u]@a1+b), H2 = leaky(H1@a2+b), s = H2@a3+b+mask,
// softmax over rows, h_out = sum a*E.
// ---------------------------------------------------------------------------
__global__ __launch_bounds__(256, 1) void attn_path_kernel(
    int mode, int Lr, int ntiles,
    const int* __restrict__ user,
    const float* __restrict__ user_embs,
    const float* __restrict__ main_embs,   // item_embs (mode0) / user_embs (mode1)
    const float* __restrict__ rate_embs,
    const int* __restrict__ gidx,          // user_hist (stride 2) / user_nbrs (stride 1)
    const __bf16* __restrict__ w_rate_pk, const float* __restrict__ b_rate,
    const __bf16* __restrict__ w_a1_pk,   const float* __restrict__ b_a1,
    const __bf16* __restrict__ w_a2_pk,   const float* __restrict__ b_a2,
    const float* __restrict__ w_a3,       const float* __restrict__ b_a3,
    float* __restrict__ h_out)
{
  extern __shared__ char smem[];
  const int rows = ntiles * 16;
  __bf16* E    = (__bf16*)smem;            // rows x 264 (stride 264 = conflict-free)
  __bf16* H1   = E + rows*264;             // rows x 264
  __bf16* Wbuf = H1 + rows*264;            // 8192 bf16 = 16KB weight slice
  __bf16* ubf  = Wbuf + 8192;              // 256 bf16 user embedding
  float* biasbuf = (float*)(ubf + 256);    // 256 f32
  float* w3buf   = biasbuf + 256;          // 256 f32
  float* scores  = w3buf + 256;            // rows f32

  const int tid = threadIdx.x;
  const int b = blockIdx.x;
  const int wave = tid >> 5, lane = tid & 31, l16 = lane & 15, lhalf = lane >> 4;
  const int ko = lhalf ? 8 : 0;
  const int npass = (ntiles > 8) ? 2 : 1;

  // ---- phase 0: user emb, biases, attn3 vector ----
  {
    int uid = user[b];
    ubf[tid] = (__bf16)user_embs[(size_t)uid*EDIM + tid];
    biasbuf[tid] = (mode == 0 ? b_rate : b_a1)[tid];
    w3buf[tid] = w_a3[tid];
  }
  __syncthreads();

  // ---- phase 1: build E ----
  if (mode == 0) {
    for (int pass = 0; pass < npass; ++pass) {
      const int t = wave + pass*8;
      const bool tv = t < ntiles;
      const int l = t*16 + l16;
      const bool rv = tv && (l < Lr);
      int idx = 0, ridx = 0;
      if (rv) { idx = gidx[(b*Lr + l)*2]; ridx = gidx[(b*Lr + l)*2 + 1]; }
      v8f c[16] = {};
      for (int kb = 0; kb < 16; ++kb) {
        stage_w(Wbuf, w_rate_pk, kb, 16, tid);
        __syncthreads();
        if (tv) {
          bf16x16 a;
          if (rv) {
            const float* src = (kb < 8)
                ? (main_embs + (size_t)idx*EDIM + kb*32 + ko)
                : (rate_embs + (size_t)ridx*EDIM + (kb-8)*32 + ko);
            cvt8(a, 0, *(const float4*)(src),    *(const float4*)(src+4));
            cvt8(a, 8, *(const float4*)(src+16), *(const float4*)(src+20));
          } else {
#pragma unroll
            for (int i = 0; i < 16; ++i) a[i] = (__bf16)0.f;
          }
          wmma_sweep16(Wbuf, lane, a, c);
        }
        __syncthreads();
      }
      if (tv) {
#pragma unroll
        for (int nt = 0; nt < 16; ++nt) {
          int n = nt*16 + l16;
          float bn = biasbuf[n];
#pragma unroll
          for (int r = 0; r < 8; ++r) {
            int row = t*16 + (lhalf ? r+8 : r);
            E[row*264 + n] = (__bf16)(c[nt][r] + bn);
          }
        }
      }
    }
  } else {
    for (int i = tid; i < rows*EDIM; i += 256) {
      int row = i >> 8, col = i & 255;
      int nb = gidx[b*Lr + row];
      E[row*264 + col] = (__bf16)main_embs[(size_t)nb*EDIM + col];
    }
  }
  __syncthreads();
  biasbuf[tid] = b_a1[tid];
  __syncthreads();

  // ---- phase 2: H1 = leaky([E|u] @ a1 + b) ----
  for (int pass = 0; pass < npass; ++pass) {
    const int t = wave + pass*8;
    const bool tv = t < ntiles;
    const int l = t*16 + l16;
    v8f c[16] = {};
    for (int kb = 0; kb < 16; ++kb) {
      stage_w(Wbuf, w_a1_pk, kb, 16, tid);
      __syncthreads();
      if (tv) {
        const __bf16* sp = (kb < 8) ? (E + l*264 + kb*32 + ko)
                                    : (ubf + (kb-8)*32 + ko);
        bf16x16 a = combine(*(const bf16x8*)sp, *(const bf16x8*)(sp+16));
        wmma_sweep16(Wbuf, lane, a, c);
      }
      __syncthreads();
    }
    if (tv) {
#pragma unroll
      for (int nt = 0; nt < 16; ++nt) {
        int n = nt*16 + l16;
        float bn = biasbuf[n];
#pragma unroll
        for (int r = 0; r < 8; ++r) {
          int row = t*16 + (lhalf ? r+8 : r);
          float v = c[nt][r] + bn;
          v = v > 0.f ? v : 0.1f*v;
          H1[row*264 + n] = (__bf16)v;
        }
      }
    }
  }
  __syncthreads();
  biasbuf[tid] = b_a2[tid];
  __syncthreads();

  // ---- phase 3+4: H2 = leaky(H1@a2+b); s = H2@a3 + b3 + mask ----
  {
    const float b3 = b_a3[0];
    for (int pass = 0; pass < npass; ++pass) {
      const int t = wave + pass*8;
      const bool tv = t < ntiles;
      const int l = t*16 + l16;
      v8f c[16] = {};
      for (int kb = 0; kb < 8; ++kb) {
        stage_w(Wbuf, w_a2_pk, kb, 8, tid);
        __syncthreads();
        if (tv) {
          const __bf16* sp = H1 + l*264 + kb*32 + ko;
          bf16x16 a = combine(*(const bf16x8*)sp, *(const bf16x8*)(sp+16));
          wmma_sweep16(Wbuf, lane, a, c);
        }
        __syncthreads();
      }
      if (tv) {
        float sp8[8];
#pragma unroll
        for (int r = 0; r < 8; ++r) sp8[r] = 0.f;
#pragma unroll
        for (int nt = 0; nt < 16; ++nt) {
          int n = nt*16 + l16;
          float bn = biasbuf[n], w3 = w3buf[n];
#pragma unroll
          for (int r = 0; r < 8; ++r) {
            float v = c[nt][r] + bn;
            v = v > 0.f ? v : 0.1f*v;
            sp8[r] += v * w3;
          }
        }
#pragma unroll
        for (int r = 0; r < 8; ++r)
#pragma unroll
          for (int m = 1; m < 16; m <<= 1)
            sp8[r] += __shfl_xor(sp8[r], m, 16);
        if (l16 == 0) {
#pragma unroll
          for (int r = 0; r < 8; ++r) {
            int row = t*16 + (lhalf ? r+8 : r);
            float sc;
            if (row < Lr) {
              int gi = (mode == 0) ? gidx[(b*Lr + row)*2] : gidx[b*Lr + row];
              sc = sp8[r] + b3 + (gi == 0 ? NEGV : 0.f);
            } else sc = NEGV;
            scores[row] = sc;
          }
        }
      }
    }
  }
  __syncthreads();

  // ---- phase 5: softmax over rows + weighted sum of E ----
  {
    float* red = (float*)Wbuf;   // Wbuf free now
    float m = (tid < rows) ? scores[tid] : -3.4e38f;
    red[tid] = m; __syncthreads();
    for (int s = 128; s > 0; s >>= 1) {
      if (tid < s) red[tid] = fmaxf(red[tid], red[tid+s]);
      __syncthreads();
    }
    float mx = red[0]; __syncthreads();
    float e = (tid < rows) ? __expf(scores[tid] - mx) : 0.f;
    if (tid < rows) scores[tid] = e;
    red[tid] = e; __syncthreads();
    for (int s = 128; s > 0; s >>= 1) {
      if (tid < s) red[tid] += red[tid+s];
      __syncthreads();
    }
    float S = red[0]; __syncthreads();
    float acc = 0.f;
    for (int l = 0; l < rows; ++l)
      acc += scores[l] * (float)E[l*264 + tid];
    h_out[(size_t)b*EDIM + tid] = acc / S;
  }
}

// ---------------------------------------------------------------------------
// Head: h = relu([h_item|h_social]@fuse+b); h2 = [h|u]@self+b;
// unexp = h2@unexp+b; exp = h2@exp+b.  64 WGs x 16-row tiles.
// ---------------------------------------------------------------------------
__global__ __launch_bounds__(256, 1) void final_mlp_kernel(
    const float* __restrict__ h_item, const float* __restrict__ h_social,
    const int* __restrict__ user, const float* __restrict__ user_embs,
    const __bf16* __restrict__ fuse_pk,  const float* __restrict__ fuse_b,
    const __bf16* __restrict__ self_pk,  const float* __restrict__ self_b,
    const __bf16* __restrict__ unexp_pk, const float* __restrict__ unexp_b,
    const __bf16* __restrict__ expl_pk,  const float* __restrict__ expl_b,
    float* __restrict__ out, int Bn)
{
  __shared__ __bf16 Abuf[16*520];
  __shared__ __bf16 ubuf[16*264];
  __shared__ __bf16 Hbuf[16*264];
  const int tid = threadIdx.x;
  const int m0 = blockIdx.x * 16;
  const int wave = tid >> 5, lane = tid & 31, l16 = lane & 15, lhalf = lane >> 4;
  const int ko = lhalf ? 8 : 0;

  for (int i = tid; i < 16*512; i += 256) {
    int row = i >> 9, col = i & 511;
    float v = (col < 256) ? h_item[(size_t)(m0+row)*EDIM + col]
                          : h_social[(size_t)(m0+row)*EDIM + (col-256)];
    Abuf[row*520 + col] = (__bf16)v;
  }
  for (int i = tid; i < 16*256; i += 256) {
    int row = i >> 8, col = i & 255;
    int u = user[m0 + row];
    ubuf[row*264 + col] = (__bf16)user_embs[(size_t)u*EDIM + col];
  }
  __syncthreads();

  // fuse (K=512)
  v8f cf[2] = {};
  for (int kb = 0; kb < 16; ++kb) {
    const __bf16* sp = Abuf + l16*520 + kb*32 + ko;
    bf16x16 a = combine(*(const bf16x8*)sp, *(const bf16x8*)(sp+16));
    bf16x16 b0 = *(const bf16x16*)(fuse_pk + (size_t)(((wave*2+0)*16 + kb)*32 + lane)*16);
    bf16x16 b1 = *(const bf16x16*)(fuse_pk + (size_t)(((wave*2+1)*16 + kb)*32 + lane)*16);
    cf[0] = wmma_bf16(a, b0, cf[0]);
    cf[1] = wmma_bf16(a, b1, cf[1]);
  }
#pragma unroll
  for (int j = 0; j < 2; ++j) {
    int n = (wave*2 + j)*16 + l16;
    float bn = fuse_b[n];
#pragma unroll
    for (int r = 0; r < 8; ++r) {
      int row = lhalf ? r+8 : r;
      float v = cf[j][r] + bn;
      Hbuf[row*264 + n] = (__bf16)(v > 0.f ? v : 0.f);
    }
  }
  __syncthreads();

  // self (K=512: [Hbuf|ubuf])
  v8f cs[2] = {};
  for (int kb = 0; kb < 16; ++kb) {
    const __bf16* sp = (kb < 8) ? (Hbuf + l16*264 + kb*32 + ko)
                                : (ubuf + l16*264 + (kb-8)*32 + ko);
    bf16x16 a = combine(*(const bf16x8*)sp, *(const bf16x8*)(sp+16));
    bf16x16 b0 = *(const bf16x16*)(self_pk + (size_t)(((wave*2+0)*16 + kb)*32 + lane)*16);
    bf16x16 b1 = *(const bf16x16*)(self_pk + (size_t)(((wave*2+1)*16 + kb)*32 + lane)*16);
    cs[0] = wmma_bf16(a, b0, cs[0]);
    cs[1] = wmma_bf16(a, b1, cs[1]);
  }
  __syncthreads();
#pragma unroll
  for (int j = 0; j < 2; ++j) {
    int n = (wave*2 + j)*16 + l16;
    float bn = self_b[n];
#pragma unroll
    for (int r = 0; r < 8; ++r) {
      int row = lhalf ? r+8 : r;
      Hbuf[row*264 + n] = (__bf16)(cs[j][r] + bn);
    }
  }
  __syncthreads();

  // unexp + exp (K=256)
  v8f cu[2] = {}, ce[2] = {};
  for (int kb = 0; kb < 8; ++kb) {
    const __bf16* sp = Hbuf + l16*264 + kb*32 + ko;
    bf16x16 a = combine(*(const bf16x8*)sp, *(const bf16x8*)(sp+16));
#pragma unroll
    for (int j = 0; j < 2; ++j) {
      int nt = wave*2 + j;
      bf16x16 bu = *(const bf16x16*)(unexp_pk + (size_t)((nt*8 + kb)*32 + lane)*16);
      bf16x16 be = *(const bf16x16*)(expl_pk  + (size_t)((nt*8 + kb)*32 + lane)*16);
      cu[j] = wmma_bf16(a, bu, cu[j]);
      ce[j] = wmma_bf16(a, be, ce[j]);
    }
  }
#pragma unroll
  for (int j = 0; j < 2; ++j) {
    int n = (wave*2 + j)*16 + l16;
    float bu = unexp_b[n], be = expl_b[n];
#pragma unroll
    for (int r = 0; r < 8; ++r) {
      int row = m0 + (lhalf ? r+8 : r);
      out[(size_t)row*EDIM + n] = cu[j][r] + bu;
      out[(size_t)Bn*EDIM + (size_t)row*EDIM + n] = ce[j][r] + be;
    }
  }
}

// ---------------------------------------------------------------------------
extern "C" void kernel_launch(void* const* d_in, const int* in_sizes, int n_in,
                              void* d_out, int out_size, void* d_ws, size_t ws_size,
                              hipStream_t stream) {
  const float* W_A1I = (const float*)d_in[0];
  const float* B_A1I = (const float*)d_in[1];
  const float* W_A2I = (const float*)d_in[2];
  const float* B_A2I = (const float*)d_in[3];
  const float* W_A3I = (const float*)d_in[4];
  const float* B_A3I = (const float*)d_in[5];
  const float* W_A1U = (const float*)d_in[6];
  const float* B_A1U = (const float*)d_in[7];
  const float* W_A2U = (const float*)d_in[8];
  const float* B_A2U = (const float*)d_in[9];
  const float* W_A3U = (const float*)d_in[10];
  const float* B_A3U = (const float*)d_in[11];
  const float* W_RATE = (const float*)d_in[12];
  const float* B_RATE = (const float*)d_in[13];
  const float* W_FUSE = (const float*)d_in[14];
  const float* B_FUSE = (const float*)d_in[15];
  const float* W_SELF = (const float*)d_in[16];
  const float* B_SELF = (const float*)d_in[17];
  const float* W_EXP  = (const float*)d_in[18];
  const float* B_EXP  = (const float*)d_in[19];
  const float* W_UNEXP = (const float*)d_in[20];
  const float* B_UNEXP = (const float*)d_in[21];
  const float* USER_EMBS = (const float*)d_in[22];
  const float* ITEM_EMBS = (const float*)d_in[23];
  const float* RATE_EMBS = (const float*)d_in[24];
  const int* USER = (const int*)d_in[25];
  const int* HIST = (const int*)d_in[26];
  const int* NBRS = (const int*)d_in[27];

  __bf16* wsb = (__bf16*)d_ws;
  __bf16* rate_pk  = wsb + PK_RATE;
  __bf16* a1i_pk   = wsb + PK_A1I;
  __bf16* a2i_pk   = wsb + PK_A2I;
  __bf16* a1u_pk   = wsb + PK_A1U;
  __bf16* a2u_pk   = wsb + PK_A2U;
  __bf16* fuse_pk  = wsb + PK_FUSE;
  __bf16* self_pk  = wsb + PK_SELF;
  __bf16* unexp_pk = wsb + PK_UNEXP;
  __bf16* expl_pk  = wsb + PK_EXP;
  float* h_item   = (float*)((char*)d_ws + WS_HITEM_BYTES);
  float* h_social = (float*)((char*)d_ws + WS_HSOC_BYTES);

  // --- pack weights into WMMA B-fragment layout (bf16) ---
  pack_w_kernel<<<512, 256, 0, stream>>>(W_RATE,  rate_pk,  512, 256);
  pack_w_kernel<<<512, 256, 0, stream>>>(W_A1I,   a1i_pk,   512, 256);
  pack_w_kernel<<<256, 256, 0, stream>>>(W_A2I,   a2i_pk,   256, 256);
  pack_w_kernel<<<512, 256, 0, stream>>>(W_A1U,   a1u_pk,   512, 256);
  pack_w_kernel<<<256, 256, 0, stream>>>(W_A2U,   a2u_pk,   256, 256);
  pack_w_kernel<<<512, 256, 0, stream>>>(W_FUSE,  fuse_pk,  512, 256);
  pack_w_kernel<<<512, 256, 0, stream>>>(W_SELF,  self_pk,  512, 256);
  pack_w_kernel<<<256, 256, 0, stream>>>(W_UNEXP, unexp_pk, 256, 256);
  pack_w_kernel<<<256, 256, 0, stream>>>(W_EXP,   expl_pk,  256, 256);

  // dynamic LDS: rows*264*2 (E) + rows*264*2 (H1) + 16384 (Wbuf) + 512 (u)
  //            + 1024 (bias) + 1024 (w3) + rows*4 (scores)
  const size_t ldsA = (size_t)208*264*2*2 + 16384 + 512 + 2048 + 208*4; // item
  const size_t ldsB = (size_t)64 *264*2*2 + 16384 + 512 + 2048 + 64*4;  // social

  // --- item attention branch (13 M-tiles of 16 over L=200) ---
  attn_path_kernel<<<BATCH, 256, ldsA, stream>>>(
      0, HISTL, 13, USER, USER_EMBS, ITEM_EMBS, RATE_EMBS, HIST,
      rate_pk, B_RATE, a1i_pk, B_A1I, a2i_pk, B_A2I, W_A3I, B_A3I, h_item);

  // --- social attention branch (4 M-tiles over N=64) ---
  attn_path_kernel<<<BATCH, 256, ldsB, stream>>>(
      1, NNBR, 4, USER, USER_EMBS, USER_EMBS, RATE_EMBS, NBRS,
      rate_pk, B_A1U, a1u_pk, B_A1U, a2u_pk, B_A2U, W_A3U, B_A3U, h_social);

  // --- head MLP ---
  final_mlp_kernel<<<BATCH/16, 256, 0, stream>>>(
      h_item, h_social, USER, USER_EMBS,
      fuse_pk, B_FUSE, self_pk, B_SELF,
      unexp_pk, B_UNEXP, expl_pk, B_EXP,
      (float*)d_out, BATCH);
}
